// CombinedLoss_32392643346804
// MI455X (gfx1250) — compile-verified, matrix operand order checked by reading, not run on hardware
//
#include <hip/hip_runtime.h>
#include <float.h>
#include <math.h>

typedef __attribute__((ext_vector_type(16))) __bf16 v16bf;
typedef __attribute__((ext_vector_type(8)))  __bf16 v8bf;
typedef __attribute__((ext_vector_type(2)))  __bf16 v2bf;
typedef __attribute__((ext_vector_type(8)))  float  v8f;

#define BN 8192
#define DN 192
#define CN 1024
#define MARGIN_F 0.2f
#define EPS_F 1e-6f

#define NSPLIT 4
#define TILES (BN / NSPLIT / 16)   // 128 N-tiles per split
#define LDS_STRIDE 200             // bf16 elems per row (192 + 8 pad) -> 100 dwords, bank-conflict free

__device__ __forceinline__ float wave_sum(float v) {
#pragma unroll
  for (int off = 16; off >= 1; off >>= 1) v += __shfl_xor(v, off, 32);
  return v;
}
__device__ __forceinline__ float wave_max(float v) {
#pragma unroll
  for (int off = 16; off >= 1; off >>= 1) v = fmaxf(v, __shfl_xor(v, off, 32));
  return v;
}

// ---------------- squared norms: one wave per row ----------------
__global__ void sq_norm_kernel(const float* __restrict__ e, float* __restrict__ sq) {
  const int wave = threadIdx.x >> 5, lane = threadIdx.x & 31;
  const int row = blockIdx.x * 8 + wave;
  const float* r = e + (long)row * DN;
  float s = 0.f;
#pragma unroll
  for (int t = 0; t < DN / 32; ++t) { float x = r[lane + 32 * t]; s += x * x; }
  s = wave_sum(s);
  if (lane == 0) sq[row] = s;
}

// ---------------- cross entropy: one wave per row ----------------
__global__ void ce_kernel(const float* __restrict__ logits, const int* __restrict__ tgt,
                          float* __restrict__ acc) {
  const int wave = threadIdx.x >> 5, lane = threadIdx.x & 31;
  const int row = blockIdx.x * 8 + wave;
  const float* r = logits + (long)row * CN;
  float m = -FLT_MAX;
#pragma unroll
  for (int t = 0; t < CN / 32; ++t) m = fmaxf(m, r[lane + 32 * t]);
  m = wave_max(m);
  float s = 0.f;
#pragma unroll
  for (int t = 0; t < CN / 32; ++t) s += expf(r[lane + 32 * t] - m);
  s = wave_sum(s);
  if (lane == 0) {
    float loss = m + logf(s) - r[tgt[row]];
    atomicAdd(acc + 0, loss);
  }
}

// ---------- cooperative staging: 16 E-rows -> LDS as split bf16 (hi/lo) ----------
// rows n0..n0+15 are contiguous in memory: one coalesced 12KB streak per tile.
__device__ __forceinline__ void stage_tile(const float* __restrict__ e,
                                           const float* __restrict__ sq,
                                           const int* __restrict__ tgt,
                                           int n0, int tid,
                                           __bf16* hiB, __bf16* loB,
                                           float* sqB, int* tgtB) {
  const float* base = e + (long)n0 * DN;
#pragma unroll
  for (int t = 0; t < (16 * DN) / (2 * 256); ++t) {   // 6 pairs per thread
    const int idx = 2 * (tid + t * 256);              // even -> pair stays in one row
    const int row = idx / DN, k = idx % DN;
    const float x0 = base[idx], x1 = base[idx + 1];
    const __bf16 h0 = (__bf16)x0, h1 = (__bf16)x1;
    v2bf hp, lp;
    hp[0] = h0; hp[1] = h1;
    lp[0] = (__bf16)(x0 - (float)h0);
    lp[1] = (__bf16)(x1 - (float)h1);
    *(v2bf*)(hiB + row * LDS_STRIDE + k) = hp;
    *(v2bf*)(loB + row * LDS_STRIDE + k) = lp;
  }
  if (tid < 16) { sqB[tid] = sq[n0 + tid]; tgtB[tid] = tgt[n0 + tid]; }
}

// join two 16B LDS segments into one WMMA operand
__device__ __forceinline__ v16bf ld_op(const __bf16* p0, const __bf16* p1) {
  union U { v16bf v; struct { v8bf a, b; } s; } u;
  u.s.a = *(const v8bf*)p0;
  u.s.b = *(const v8bf*)p1;
  return u.v;
}

// ------ fused Gram (split-bf16 WMMA) + hard positive/negative mining ------
// one wave per 16-row M-tile; block shares LDS-staged B tiles; N range split 4 ways.
// Three independent WMMA accumulator chains (hi*hi, hi*lo, lo*hi) keep the matrix
// pipe free of back-to-back D->C RAW stalls.
__global__ __launch_bounds__(256, 1)
void mine_kernel(const float* __restrict__ e, const float* __restrict__ sq,
                 const int* __restrict__ tgt,
                 float* __restrict__ posV, int* __restrict__ posIp,
                 float* __restrict__ negV, int* __restrict__ negIp) {
  __shared__ __align__(16) __bf16 shHi[2][16 * LDS_STRIDE];
  __shared__ __align__(16) __bf16 shLo[2][16 * LDS_STRIDE];
  __shared__ float shSq[2][16];
  __shared__ int   shTg[2][16];

  const int tid   = threadIdx.x;
  const int wave  = tid >> 5;
  const int lane  = tid & 31;
  const int half  = lane >> 4;
  const int ln    = lane & 15;
  const int split = blockIdx.x % NSPLIT;
  const int m0    = ((blockIdx.x / NSPLIT) * 8 + wave) * 16;
  const int nBase = split * (BN / NSPLIT);

  // A tile (rows m0..m0+15) resident in VGPRs, split into bf16 hi/lo, WMMA A layout:
  // lane holds row M=ln; elems 0..7 = K half*8+0..7, elems 8..15 = K 16+half*8+0..7 per chunk.
  const float* arow = e + (long)(m0 + ln) * DN;
  v16bf ahi[6], alo[6];
#pragma unroll
  for (int c = 0; c < 6; ++c) {
    const int k0 = 32 * c;
#pragma unroll
    for (int j = 0; j < 8; ++j) {
      float x = arow[k0 + half * 8 + j];
      __bf16 h = (__bf16)x;
      ahi[c][j] = h; alo[c][j] = (__bf16)(x - (float)h);
      float y = arow[k0 + 16 + half * 8 + j];
      __bf16 h2 = (__bf16)y;
      ahi[c][j + 8] = h2; alo[c][j + 8] = (__bf16)(y - (float)h2);
    }
  }

  float sqRow[8];
  int   tgtRow[8];
#pragma unroll
  for (int r = 0; r < 8; ++r) {
    const int i = m0 + r + 8 * half;
    sqRow[r] = sq[i];
    tgtRow[r] = tgt[i];
  }

  float posB[8], negB[8];
  int   posI[8], negI[8];
#pragma unroll
  for (int r = 0; r < 8; ++r) { posB[r] = -1.f; negB[r] = FLT_MAX; posI[r] = 0; negI[r] = 0; }

  stage_tile(e, sq, tgt, nBase, tid, shHi[0], shLo[0], shSq[0], shTg[0]);
  __syncthreads();

#pragma unroll 1
  for (int t = 0; t < TILES; ++t) {
    const int cur = t & 1;
    // prefetch next tile into the other buffer; overlaps with WMMA below
    if (t + 1 < TILES)
      stage_tile(e, sq, tgt, nBase + (t + 1) * 16, tid,
                 shHi[1 - cur], shLo[1 - cur], shSq[1 - cur], shTg[1 - cur]);

    const __bf16* hiB = shHi[cur];
    const __bf16* loB = shLo[cur];
    const int rowOff = ln * LDS_STRIDE;

    v8f accHH = {0.f, 0.f, 0.f, 0.f, 0.f, 0.f, 0.f, 0.f};
    v8f accHL = {0.f, 0.f, 0.f, 0.f, 0.f, 0.f, 0.f, 0.f};
    v8f accLH = {0.f, 0.f, 0.f, 0.f, 0.f, 0.f, 0.f, 0.f};
#pragma unroll
    for (int c = 0; c < 6; ++c) {
      const int o0 = rowOff + 32 * c + half * 8;
      const int o1 = o0 + 16;
      const v16bf bh = ld_op(hiB + o0, hiB + o1);
      const v16bf bl = ld_op(loB + o0, loB + o1);
      accHH = __builtin_amdgcn_wmma_f32_16x16x32_bf16(false, ahi[c], false, bh,
                                                      (short)0, accHH, false, false);
      accHL = __builtin_amdgcn_wmma_f32_16x16x32_bf16(false, ahi[c], false, bl,
                                                      (short)0, accHL, false, false);
      accLH = __builtin_amdgcn_wmma_f32_16x16x32_bf16(false, alo[c], false, bh,
                                                      (short)0, accLH, false, false);
    }

    const int   n0     = nBase + t * 16;
    const int   j      = n0 + ln;            // this lane's column for all 8 C rows
    const float sqCol  = shSq[cur][ln];
    const int   tgtCol = shTg[cur][ln];
#pragma unroll
    for (int r = 0; r < 8; ++r) {
      const int i = m0 + r + 8 * half;
      const float dot = accHH[r] + (accHL[r] + accLH[r]);   // small terms first
      const float d2 = sqRow[r] + sqCol - 2.0f * dot;
      const float dist = sqrtf(fmaxf(d2, 0.f));
      const bool same = (tgtRow[r] == tgtCol);
      if (same && (i != j)) {
        if (dist > posB[r]) { posB[r] = dist; posI[r] = j; }   // strict > keeps first index
      }
      if (!same) {
        if (dist < negB[r]) { negB[r] = dist; negI[r] = j; }
      }
    }
    __syncthreads();
  }

  // merge across the 16 lanes that hold the same rows (xor masks stay within halves)
#pragma unroll
  for (int r = 0; r < 8; ++r) {
#pragma unroll
    for (int off = 1; off < 16; off <<= 1) {
      float ob = __shfl_xor(posB[r], off, 32);
      int   oi = __shfl_xor(posI[r], off, 32);
      if (ob > posB[r] || (ob == posB[r] && oi < posI[r])) { posB[r] = ob; posI[r] = oi; }
      float nb = __shfl_xor(negB[r], off, 32);
      int   ni = __shfl_xor(negI[r], off, 32);
      if (nb < negB[r] || (nb == negB[r] && ni < negI[r])) { negB[r] = nb; negI[r] = ni; }
    }
  }

  if (ln == 0) {
#pragma unroll
    for (int r = 0; r < 8; ++r) {
      const int i = m0 + r + 8 * half;
      posV[split * BN + i] = posB[r];
      posIp[split * BN + i] = posI[r];
      negV[split * BN + i] = negB[r];
      negIp[split * BN + i] = negI[r];
    }
  }
}

// ---------------- merge the NSPLIT partial argmax/argmin results ----------------
__global__ void merge_kernel(const float* __restrict__ posV, const int* __restrict__ posIp,
                             const float* __restrict__ negV, const int* __restrict__ negIp,
                             int* __restrict__ posIdx, int* __restrict__ negIdx,
                             int* __restrict__ validF) {
  const int i = blockIdx.x * 256 + threadIdx.x;
  float pb = -1.f;     int pi = 0;
  float nb = FLT_MAX;  int ni = 0;
#pragma unroll
  for (int s = 0; s < NSPLIT; ++s) {
    const float v = posV[s * BN + i]; const int ix = posIp[s * BN + i];
    if (v > pb || (v == pb && ix < pi)) { pb = v; pi = ix; }
    const float w = negV[s * BN + i]; const int jx = negIp[s * BN + i];
    if (w < nb || (w == nb && jx < ni)) { nb = w; ni = jx; }
  }
  posIdx[i] = pi;
  negIdx[i] = ni;
  validF[i] = (pb >= 0.f && nb < FLT_MAX) ? 1 : 0;
}

// ---------------- triplet margin loss in exact fp32: one wave per row ----------------
__global__ void triplet_kernel(const float* __restrict__ e, const int* __restrict__ posIdx,
                               const int* __restrict__ negIdx, const int* __restrict__ validF,
                               float* __restrict__ acc) {
  const int wave = threadIdx.x >> 5, lane = threadIdx.x & 31;
  const int row = blockIdx.x * 8 + wave;
  const float* a = e + (long)row * DN;
  const float* p = e + (long)posIdx[row] * DN;
  const float* n = e + (long)negIdx[row] * DN;
  float sap = 0.f, san = 0.f;
#pragma unroll
  for (int t = 0; t < DN / 32; ++t) {
    const int k = lane + 32 * t;
    const float dp = a[k] - p[k] + EPS_F;
    const float dn = a[k] - n[k] + EPS_F;
    sap += dp * dp;
    san += dn * dn;
  }
  sap = wave_sum(sap);
  san = wave_sum(san);
  if (lane == 0) {
    const float per = fmaxf(sqrtf(sap) - sqrtf(san) + MARGIN_F, 0.f);
    const float w = validF[row] ? 1.f : 0.f;
    atomicAdd(acc + 1, per * w);
    atomicAdd(acc + 2, w);
  }
}

__global__ void finalize_kernel(const float* __restrict__ acc, float* __restrict__ out) {
  const float cls = acc[0] / (float)BN;
  const float trip = (acc[2] > 0.f) ? (acc[1] / fmaxf(acc[2], 1.f)) : 0.f;
  out[0] = cls + trip;
}

extern "C" void kernel_launch(void* const* d_in, const int* in_sizes, int n_in,
                              void* d_out, int out_size, void* d_ws, size_t ws_size,
                              hipStream_t stream) {
  const float* emb    = (const float*)d_in[0];
  const float* logits = (const float*)d_in[1];
  const int*   tgt    = (const int*)d_in[2];
  float* out = (float*)d_out;

  float* sqn    = (float*)d_ws;                 // [BN]
  float* posV   = sqn + BN;                     // [NSPLIT*BN]
  int*   posIp  = (int*)(posV + NSPLIT * BN);   // [NSPLIT*BN]
  float* negV   = (float*)(posIp + NSPLIT * BN);// [NSPLIT*BN]
  int*   negIp  = (int*)(negV + NSPLIT * BN);   // [NSPLIT*BN]
  int*   posIdx = negIp + NSPLIT * BN;          // [BN]
  int*   negIdx = posIdx + BN;                  // [BN]
  int*   validF = negIdx + BN;                  // [BN]
  float* acc    = (float*)(validF + BN);        // [3]: cls_sum, trip_sum, trip_cnt

  hipMemsetAsync(acc, 0, 3 * sizeof(float), stream);

  sq_norm_kernel<<<BN / 8, 256, 0, stream>>>(emb, sqn);
  ce_kernel<<<BN / 8, 256, 0, stream>>>(logits, tgt, acc);
  mine_kernel<<<(BN / 16 / 8) * NSPLIT, 256, 0, stream>>>(emb, sqn, tgt,
                                                          posV, posIp, negV, negIp);
  merge_kernel<<<BN / 256, 256, 0, stream>>>(posV, posIp, negV, negIp,
                                             posIdx, negIdx, validF);
  triplet_kernel<<<BN / 8, 256, 0, stream>>>(emb, posIdx, negIdx, validF, acc);
  finalize_kernel<<<1, 1, 0, stream>>>(acc, out);
}
